// JaxGRU_22256520528302
// MI455X (gfx1250) — matrix operand order repaired and checked by reading, not compiled, not run on hardware
//
#include <hip/hip_runtime.h>
#include <hip/hip_bf16.h>
#include <math.h>

// ---------------------------------------------------------------------------
// Problem constants (from reference)
// ---------------------------------------------------------------------------
static constexpr int BS       = 256;
static constexpr int HIST_LEN = 250;
static constexpr int FEAT     = 32;
static constexpr int KIN      = HIST_LEN * FEAT;   // 8000
static constexpr int T_       = 100;
static constexpr int ACT_DIM  = 32;
static constexpr int HID      = 1024;
static constexpr int OUT_     = 64;

// Try the CDNA5 async global->LDS copy path (inline asm). Set to 0 to fall
// back to load+ds_store staging if the assembler rejects the mnemonic.
#define USE_ASYNC_LDS 1

// ---------------------------------------------------------------------------
// WMMA types / helpers (gfx1250, wave32)
// ---------------------------------------------------------------------------
typedef __attribute__((ext_vector_type(16))) __bf16 v16bf;
typedef __attribute__((ext_vector_type(8)))  float  v8f;

__device__ __forceinline__ v8f wmma_bf16(v16bf a, v16bf b, v8f c) {
    // 8 args: (neg_a, A, neg_b, B, c_mod, C, reuse_a, reuse_b)
    return __builtin_amdgcn_wmma_f32_16x16x32_bf16(false, a, false, b,
                                                   (short)0, c, false, false);
}

// A matrix fragment, 16x32 bf16, row-major source with leading dim lda.
// ISA layout: lanes 0-15 hold M=lane; lanes 16-31 hold M=lane-16.
//   grp=lane>>4: elements 0..7  <- K = kb + 8*grp + (0..7)
//                elements 8..15 <- K = kb + 16 + 8*grp + (0..7)
__device__ __forceinline__ v16bf load_a_frag(const __bf16* A, size_t lda,
                                             int m0, int kb, int lane) {
    int grp = (lane >> 4) & 1;
    int m   = lane & 15;
    const __bf16* p = A + (size_t)(m0 + m) * lda + (size_t)kb;
    union { uint4 u[2]; v16bf v; } u;
    u.u[0] = *(const uint4*)(p + 8 * grp);
    u.u[1] = *(const uint4*)(p + 16 + 8 * grp);
    return u.v;
}

// Same fragment sourced from an LDS tile (16 rows x HID bf16, row-major).
__device__ __forceinline__ v16bf load_a_frag_lds(const __bf16* S, int kb, int lane) {
    int grp = (lane >> 4) & 1;
    int m   = lane & 15;
    const __bf16* p = S + (size_t)m * HID + (size_t)kb;
    union { uint4 u[2]; v16bf v; } u;
    u.u[0] = *(const uint4*)(p + 8 * grp);       // ds_load_b128
    u.u[1] = *(const uint4*)(p + 16 + 8 * grp);  // ds_load_b128
    return u.v;
}

// B matrix fragment, 32x16 bf16 (KxN), sourced from a TRANSPOSED weight
// matrix Bt (N x K, row-major, leading dim ldb) so the per-lane data is
// contiguous. Lane n=lane&15; grp=lane>>4: elements 0..7 <- K=8*grp+(0..7), 8..15 <- K=16+8*grp+(0..7).
__device__ __forceinline__ v16bf load_b_frag(const __bf16* Bt, size_t ldb,
                                             int n0, int kb, int lane) {
    int grp = (lane >> 4) & 1;
    int n   = lane & 15;
    const __bf16* p = Bt + (size_t)(n0 + n) * ldb + (size_t)kb;
    union { uint4 u[2]; v16bf v; } u;
    u.u[0] = *(const uint4*)(p + 8 * grp);
    u.u[1] = *(const uint4*)(p + 16 + 8 * grp);
    return u.v;
}

__device__ __forceinline__ float sigmoidf_(float x) {
    return 1.0f / (1.0f + expf(-x));
}

// ---------------------------------------------------------------------------
// Conversion kernels (one-time prep)
// ---------------------------------------------------------------------------
__global__ void k_convert(const float* __restrict__ in,
                          __bf16* __restrict__ out, size_t n) {
    size_t i = (size_t)blockIdx.x * blockDim.x + threadIdx.x;
    if (i < n) out[i] = (__bf16)in[i];
}

// in: K x N (row-major) fp32  ->  out: N x K (row-major) bf16
__global__ void k_convert_transpose(const float* __restrict__ in,
                                    __bf16* __restrict__ out, int K, int N) {
    size_t i = (size_t)blockIdx.x * blockDim.x + threadIdx.x;
    size_t total = (size_t)K * (size_t)N;
    if (i >= total) return;
    size_t k = i / (size_t)N;
    size_t n = i % (size_t)N;
    out[n * (size_t)K + k] = (__bf16)in[i];
}

// ---------------------------------------------------------------------------
// Generic WMMA GEMM: C[M,N] = act(A[M,K] * Bt[N,K]^T + bias), one wave per
// 16x16 output tile, software-pipelined K loop. Optionally writes a bf16
// copy of C.
// ---------------------------------------------------------------------------
template <bool RELU, bool WRITE_BF16>
__global__ void k_gemm_bf16(const __bf16* __restrict__ A, size_t lda,
                            const __bf16* __restrict__ Bt, size_t ldb,
                            const float* __restrict__ bias,
                            float* __restrict__ C, size_t ldc,
                            __bf16* __restrict__ Cb, size_t ldcb,
                            int M, int N, int K) {
    const int lane = threadIdx.x & 31;
    const int wave = blockIdx.x * (blockDim.x >> 5) + (threadIdx.x >> 5);
    const int nTilesN = N >> 4;
    const int nTiles  = (M >> 4) * nTilesN;
    if (wave >= nTiles) return;          // uniform per wave: EXEC stays all-1s
    const int m0 = (wave / nTilesN) << 4;
    const int n0 = (wave % nTilesN) << 4;

    v8f acc = {};
    // software pipeline: fragments for kb are loaded at kb-32
    v16bf a0 = load_a_frag(A, lda, m0, 0, lane);
    v16bf b0 = load_b_frag(Bt, ldb, n0, 0, lane);
    for (int kb = 0; kb < K; kb += 32) {
        v16bf a1 = a0, b1 = b0;
        int kn = kb + 32;
        if (kn < K) {
            a1 = load_a_frag(A, lda, m0, kn, lane);
            b1 = load_b_frag(Bt, ldb, n0, kn, lane);
        }
        acc = wmma_bf16(a0, b0, acc);
        a0 = a1; b0 = b1;
    }

    const int grp = lane >> 4;
    const int col = n0 + (lane & 15);
    const float bv = bias ? bias[col] : 0.0f;
#pragma unroll
    for (int r = 0; r < 8; ++r) {
        int row = m0 + (grp << 3) + r;
        float v = acc[r] + bv;
        if (RELU) v = fmaxf(v, 0.0f);
        C[(size_t)row * ldc + col] = v;
        if (WRITE_BF16) Cb[(size_t)row * ldcb + col] = (__bf16)v;
    }
}

// ---------------------------------------------------------------------------
// Fused GRU step. Blocks of 8 waves; all 8 waves share the same batch tile
// mb, so the 16x1024 bf16 h_{t-1} tile (32 KB) is staged into LDS once per
// block via async global->LDS copies, then A fragments come from LDS.
//   gh = h_{t-1} @ Wh ; gi = a_t @ Wi + bi
//   r = sig(gi_r + gh_r); z = sig(gi_z + gh_z)
//   n = tanh(gi_n + r*(gh_n + bhn)) ; h = (1-z)*n + z*h_old
// ---------------------------------------------------------------------------
__global__ void k_gru_step(const __bf16* __restrict__ hprev_b, size_t lda_h,
                           const float*  __restrict__ hprev_f,
                           const __bf16* __restrict__ Aact, size_t lda_a,
                           const __bf16* __restrict__ WhT,   // 3072 x 1024
                           const __bf16* __restrict__ WiT,   // 3072 x 32
                           const float*  __restrict__ bi,    // 3072
                           const float*  __restrict__ bhn,   // 1024
                           float*  __restrict__ hnext_f,
                           __bf16* __restrict__ Hall,
                           int t) {
    __shared__ __align__(16) __bf16 smemA[16 * HID];   // 32 KB h tile

    const int lane = threadIdx.x & 31;
    const int wave = blockIdx.x * (blockDim.x >> 5) + (threadIdx.x >> 5);
    // 16 tiles over batch (256), 64 tiles over hidden (1024) => 1024 waves;
    // the 8 waves of a block share mb (wave>>6 constant within a block).
    const int mb = (wave >> 6) << 4;
    const int jb = (wave & 63) << 4;

    // ---- stage h tile (rows mb..mb+15, 2048 B each) into LDS -------------
    {
        const int tid = threadIdx.x;              // 256 threads
#pragma unroll
        for (int i = 0; i < 8; ++i) {             // 2048 chunks of 16 B
            int c    = tid + (i << 8);
            int row  = c >> 7;                    // 128 chunks per row
            int col8 = (c & 127) << 3;            // bf16 element offset
            const __bf16* gsrc = hprev_b + (size_t)(mb + row) * lda_h + col8;
            __bf16* ldst = &smemA[(size_t)row * HID + col8];
#if USE_ASYNC_LDS
            unsigned  loff  = (unsigned)(size_t)ldst;             // LDS addr[31:0]
            unsigned long long gaddr = (unsigned long long)(size_t)gsrc;
            asm volatile("global_load_async_to_lds_b128 %0, %1, off"
                         :: "v"(loff), "v"(gaddr) : "memory");
#else
            *(uint4*)ldst = *(const uint4*)gsrc;  // load + ds_store_b128
#endif
        }
#if USE_ASYNC_LDS
        asm volatile("s_wait_asynccnt 0x0" ::: "memory");
#endif
        __syncthreads();
    }

    v8f acc_r = {}, acc_z = {}, acc_gn = {}, acc_in = {};

    // ---- K loop, software-pipelined B fragments (Wh is L2-resident) ------
    v16bf br0 = load_b_frag(WhT, HID, jb,           0, lane);
    v16bf bz0 = load_b_frag(WhT, HID, HID + jb,     0, lane);
    v16bf bn0 = load_b_frag(WhT, HID, 2 * HID + jb, 0, lane);
    for (int kb = 0; kb < HID; kb += 32) {
        v16bf br1 = br0, bz1 = bz0, bn1 = bn0;
        int kn = kb + 32;
        if (kn < HID) {
            br1 = load_b_frag(WhT, HID, jb,           kn, lane);
            bz1 = load_b_frag(WhT, HID, HID + jb,     kn, lane);
            bn1 = load_b_frag(WhT, HID, 2 * HID + jb, kn, lane);
            __builtin_prefetch(WhT + (size_t)jb * HID + kn, 0, 1);
        }
        v16bf a = load_a_frag_lds(smemA, kb, lane);
        acc_r  = wmma_bf16(a, br0, acc_r);
        acc_z  = wmma_bf16(a, bz0, acc_z);
        acc_gn = wmma_bf16(a, bn0, acc_gn);
        br0 = br1; bz0 = bz1; bn0 = bn1;
    }

    // ---- gi = a_t @ Wi (K = 32, single WMMA per gate) --------------------
    {
        v16bf a2 = load_a_frag(Aact, lda_a, mb, 0, lane);
        v16bf br = load_b_frag(WiT, ACT_DIM, jb,           0, lane);
        v16bf bz = load_b_frag(WiT, ACT_DIM, HID + jb,     0, lane);
        v16bf bn = load_b_frag(WiT, ACT_DIM, 2 * HID + jb, 0, lane);
        acc_r  = wmma_bf16(a2, br, acc_r);
        acc_z  = wmma_bf16(a2, bz, acc_z);
        acc_in = wmma_bf16(a2, bn, acc_in);
    }

    const int grp = lane >> 4;
    const int j   = jb + (lane & 15);
    const float bir = bi[j];
    const float biz = bi[HID + j];
    const float bin = bi[2 * HID + j];
    const float bh  = bhn[j];

#pragma unroll
    for (int r = 0; r < 8; ++r) {
        int b = mb + (grp << 3) + r;
        float rg = sigmoidf_(acc_r[r] + bir);
        float zg = sigmoidf_(acc_z[r] + biz);
        float ng = tanhf(acc_in[r] + bin + rg * (acc_gn[r] + bh));
        float ho = hprev_f[(size_t)b * HID + j];
        float hn = (1.0f - zg) * ng + zg * ho;
        hnext_f[(size_t)b * HID + j] = hn;
        Hall[((size_t)b * T_ + t) * HID + j] = (__bf16)hn;
    }
}

// ---------------------------------------------------------------------------
// Host-side launch
// ---------------------------------------------------------------------------
static inline size_t align256(size_t x) { return (x + 255) & ~(size_t)255; }

extern "C" void kernel_launch(void* const* d_in, const int* in_sizes, int n_in,
                              void* d_out, int out_size, void* d_ws, size_t ws_size,
                              hipStream_t stream) {
    (void)in_sizes; (void)n_in; (void)out_size; (void)ws_size;

    const float* history = (const float*)d_in[0];  // (256, 250, 32)
    const float* action  = (const float*)d_in[1];  // (256, 100, 32)
    const float* W_in    = (const float*)d_in[2];  // (8000, 1024)
    const float* b_in    = (const float*)d_in[3];  // (1024,)
    const float* Wi      = (const float*)d_in[4];  // (32, 3072)
    const float* bi      = (const float*)d_in[5];  // (3072,)
    const float* Wh      = (const float*)d_in[6];  // (1024, 3072)
    const float* bhn     = (const float*)d_in[7];  // (1024,)
    const float* Wo      = (const float*)d_in[8];  // (1024, 64)
    const float* bo      = (const float*)d_in[9];  // (64,)
    float* outp          = (float*)d_out;          // (256, 100, 64)

    // Workspace layout
    char* ws = (char*)d_ws;
    size_t off = 0;
    __bf16* WinT  = (__bf16*)(ws + off); off = align256(off + (size_t)HID * KIN * 2);        // 1024 x 8000
    __bf16* WiT   = (__bf16*)(ws + off); off = align256(off + (size_t)3 * HID * ACT_DIM * 2);// 3072 x 32
    __bf16* WhT   = (__bf16*)(ws + off); off = align256(off + (size_t)3 * HID * HID * 2);    // 3072 x 1024
    __bf16* WoT   = (__bf16*)(ws + off); off = align256(off + (size_t)OUT_ * HID * 2);       // 64 x 1024
    __bf16* histb = (__bf16*)(ws + off); off = align256(off + (size_t)BS * KIN * 2);         // 256 x 8000
    __bf16* actb  = (__bf16*)(ws + off); off = align256(off + (size_t)BS * T_ * ACT_DIM * 2);// 256 x 3200
    __bf16* h0b   = (__bf16*)(ws + off); off = align256(off + (size_t)BS * HID * 2);         // 256 x 1024
    float*  ping0 = (float*) (ws + off); off = align256(off + (size_t)BS * HID * 4);
    float*  ping1 = (float*) (ws + off); off = align256(off + (size_t)BS * HID * 4);
    __bf16* Hall  = (__bf16*)(ws + off); off = align256(off + (size_t)BS * T_ * HID * 2);    // 25600 x 1024

    const int CT = 256;
    auto blocks = [](size_t n, int bs) { return (int)((n + bs - 1) / bs); };

    // --- 1) weight conversion + transpose to bf16 (N x K) -----------------
    k_convert_transpose<<<blocks((size_t)KIN * HID, CT), CT, 0, stream>>>(W_in, WinT, KIN, HID);
    k_convert_transpose<<<blocks((size_t)ACT_DIM * 3 * HID, CT), CT, 0, stream>>>(Wi, WiT, ACT_DIM, 3 * HID);
    k_convert_transpose<<<blocks((size_t)HID * 3 * HID, CT), CT, 0, stream>>>(Wh, WhT, HID, 3 * HID);
    k_convert_transpose<<<blocks((size_t)HID * OUT_, CT), CT, 0, stream>>>(Wo, WoT, HID, OUT_);
    k_convert<<<blocks((size_t)BS * KIN, CT), CT, 0, stream>>>(history, histb, (size_t)BS * KIN);
    k_convert<<<blocks((size_t)BS * T_ * ACT_DIM, CT), CT, 0, stream>>>(action, actb, (size_t)BS * T_ * ACT_DIM);

    // --- 2) h0 = relu(x @ W_in + b_in) ------------------------------------
    {
        int waves = (BS / 16) * (HID / 16);           // 1024
        k_gemm_bf16<true, true><<<blocks((size_t)waves * 32, CT), CT, 0, stream>>>(
            histb, (size_t)KIN, WinT, (size_t)KIN, b_in,
            ping0, (size_t)HID, h0b, (size_t)HID,
            BS, HID, KIN);
    }

    // --- 3) GRU recurrence -------------------------------------------------
    {
        int waves = (BS / 16) * (HID / 16);           // 1024 waves -> 128 blocks
        int nblk = blocks((size_t)waves * 32, CT);
        for (int t = 0; t < T_; ++t) {
            const __bf16* hpb = (t == 0) ? h0b : (Hall + (size_t)(t - 1) * HID);
            size_t lda_h      = (t == 0) ? (size_t)HID : (size_t)T_ * HID;
            const float* hpf  = (t & 1) ? ping1 : ping0;
            float* hnf        = (t & 1) ? ping0 : ping1;
            const __bf16* Aact = actb + (size_t)t * ACT_DIM;   // row stride T_*ACT_DIM
            k_gru_step<<<nblk, CT, 0, stream>>>(
                hpb, lda_h, hpf, Aact, (size_t)T_ * ACT_DIM,
                WhT, WiT, bi, bhn, hnf, Hall, t);
        }
    }

    // --- 4) out = Hall(25600 x 1024) @ Wo + bo -> d_out -------------------
    {
        int M = BS * T_;                              // 25600
        int waves = (M / 16) * (OUT_ / 16);           // 6400
        k_gemm_bf16<false, false><<<blocks((size_t)waves * 32, CT), CT, 0, stream>>>(
            Hall, (size_t)HID, WoT, (size_t)HID, bo,
            outp, (size_t)OUT_, (__bf16*)nullptr, (size_t)0,
            M, OUT_, HID);
    }
}
